// FlashAttention_85169201480425
// MI455X (gfx1250) — matrix operand order repaired and checked by reading, not compiled, not run on hardware
//
#include <hip/hip_runtime.h>
#include <hip/hip_bf16.h>

#define D_MODEL 1024
#define NHEADS  16
#define HD      64
#define CHUNK   256
#define GBLK    32      // 8192 / 256
#define SEQ     8192
#define BATCH   4

typedef __attribute__((ext_vector_type(16))) _Float16 v16h;
typedef __attribute__((ext_vector_type(8)))  _Float16 v8h;
typedef __attribute__((ext_vector_type(8)))  float    v8f;
typedef __attribute__((ext_vector_type(4)))  float    v4f;
typedef __attribute__((ext_vector_type(4)))  int      v4i;

union AB { v16h v; v8h h[2]; };

// ---------------------------------------------------------------------------
// CDNA5 async global->LDS copy (ASYNCcnt-tracked), with graceful fallback.
// Builtin signature (from hipcc diagnostic): (v4i AS1*, v4i AS3*, imm, imm).
// ---------------------------------------------------------------------------
#if defined(__has_builtin)
#if __has_builtin(__builtin_amdgcn_global_load_async_to_lds_b128)
#define HAS_ASYNC_LDS 1
#endif
#endif
#ifndef HAS_ASYNC_LDS
#define HAS_ASYNC_LDS 0
#endif

#if HAS_ASYNC_LDS
#if defined(__has_builtin) && __has_builtin(__builtin_amdgcn_s_wait_asynccnt)
#define WAIT_ASYNC(n) __builtin_amdgcn_s_wait_asynccnt(n)
#else
#define WAIT_ASYNC(n) asm volatile("s_wait_asynccnt %0" ::"i"(n) : "memory")
#endif
#else
#define WAIT_ASYNC(n) ((void)0)
#endif

typedef __attribute__((address_space(1))) v4i g_v4i;   // global (printed __device__)
typedef __attribute__((address_space(3))) v4i l_v4i;   // LDS    (printed __shared__)

__device__ __forceinline__ void async_copy16(const _Float16* g, _Float16* l) {
#if HAS_ASYNC_LDS
  // Generic LDS addr low 32 bits == LDS offset per the aperture mapping,
  // and generic global addr == AS1 addr, so integer casts are exact.
  __builtin_amdgcn_global_load_async_to_lds_b128(
      (g_v4i*)(unsigned long long)(const void*)g,
      (l_v4i*)(unsigned int)(unsigned long long)(void*)l,
      0, 0);
#else
  *(v8h*)l = *(const v8h*)g;
#endif
}

__device__ __forceinline__ v8f wmma_f16(v16h a, v16h b, v8f c) {
  // D = A(16x32 f16) * B(32x16 f16) + C(16x16 f32)
  return __builtin_amdgcn_wmma_f32_16x16x32_f16(false, a, false, b, (short)0, c,
                                                false, false);
}

__device__ __forceinline__ v8f zero8() {
  v8f z = {0.f, 0.f, 0.f, 0.f, 0.f, 0.f, 0.f, 0.f};
  return z;
}

// ---------------------------------------------------------------------------
// fp32 -> f16 bulk convert (x -> xh), 8 elements/thread
// ---------------------------------------------------------------------------
__global__ __launch_bounds__(256) void convert_f16(const float* __restrict__ src,
                                                   _Float16* __restrict__ dst) {
  const size_t i = (size_t)blockIdx.x * 256 + threadIdx.x;
  const v4f* s = (const v4f*)src;
  v4f a = s[i * 2 + 0];
  v4f b = s[i * 2 + 1];
  v8h o;
  o[0] = (_Float16)a[0]; o[1] = (_Float16)a[1];
  o[2] = (_Float16)a[2]; o[3] = (_Float16)a[3];
  o[4] = (_Float16)b[0]; o[5] = (_Float16)b[1];
  o[6] = (_Float16)b[2]; o[7] = (_Float16)b[3];
  ((v8h*)dst)[i] = o;
}

// ---------------------------------------------------------------------------
// Weight transpose + fp32 -> f16 convert:  WT[n][k] = W[k][n]
// ---------------------------------------------------------------------------
__global__ __launch_bounds__(256) void transpose_f16(const float* __restrict__ W,
                                                     _Float16* __restrict__ WT) {
  __shared__ float tile[32][33];
  const int bx = blockIdx.x * 32, by = blockIdx.y * 32;
  const int tx = threadIdx.x, ty = threadIdx.y;
#pragma unroll
  for (int j = 0; j < 32; j += 8)
    tile[ty + j][tx] = W[(size_t)(by + ty + j) * D_MODEL + (bx + tx)];
  __syncthreads();
#pragma unroll
  for (int j = 0; j < 32; j += 8)
    WT[(size_t)(bx + ty + j) * D_MODEL + (by + tx)] = (_Float16)tile[tx][ty + j];
}

// ---------------------------------------------------------------------------
// WMMA GEMM: Y = A(32768 x 1024, f16) @ W(1024 x 1024) + bias
//   MODE 0: write f16 [B,H,G,C,hd]           (Q and K layout)
//   MODE 2: write f16 [B,H,G,hd,C]           (V transposed)
//   MODE 3: write f32 row-major [B*N, D]     (final output)
// Block tile 128x128, 8 waves, each wave 32(M) x 64(N).
// Double-buffered LDS staged with async global->LDS copies.
// ---------------------------------------------------------------------------
template <int MODE>
__global__ __launch_bounds__(256) void wmma_gemm(const _Float16* __restrict__ A,
                                                 const _Float16* __restrict__ WT,
                                                 const float* __restrict__ bias,
                                                 void* __restrict__ Out) {
  __shared__ _Float16 sA[2][128 * 32];
  __shared__ _Float16 sB[2][128 * 32];

  const int tid  = threadIdx.x;
  const int lane = tid & 31;
  const int w    = tid >> 5;
  const int wm   = w >> 1;          // 0..3  (M direction)
  const int wn   = w & 1;           // 0..1  (N direction)
  const int hi   = lane >> 4;       // half-wave select
  const int l16  = lane & 15;

  const int m0 = blockIdx.y * 128;
  const int n0 = blockIdx.x * 128;

  const int lrow = tid >> 1;            // 0..127
  const int lcol = (tid & 1) * 16;      // 0 or 16
  const int soff = lrow * 32 + lcol;

  const _Float16* gA = A  + (size_t)(m0 + lrow) * D_MODEL + lcol;
  const _Float16* gB = WT + (size_t)(n0 + lrow) * D_MODEL + lcol;

  // prologue: stage k-step 0 into buffer 0
  async_copy16(gA + 0, &sA[0][soff]);
  async_copy16(gA + 8, &sA[0][soff + 8]);
  async_copy16(gB + 0, &sB[0][soff]);
  async_copy16(gB + 8, &sB[0][soff + 8]);

  v8f acc[2][4];
#pragma unroll
  for (int t = 0; t < 2; ++t)
#pragma unroll
    for (int u = 0; u < 4; ++u) acc[t][u] = zero8();

  for (int k0 = 0; k0 < D_MODEL; k0 += 32) {
    const int cur = (k0 >> 5) & 1;
    const int nxt = cur ^ 1;

    if (k0 + 32 < D_MODEL) {
      // issue next-stage copies, then wait for current stage (in-order retire)
      const _Float16* ga = gA + k0 + 32;
      const _Float16* gb = gB + k0 + 32;
      async_copy16(ga + 0, &sA[nxt][soff]);
      async_copy16(ga + 8, &sA[nxt][soff + 8]);
      async_copy16(gb + 0, &sB[nxt][soff]);
      async_copy16(gb + 8, &sB[nxt][soff + 8]);
      WAIT_ASYNC(4);
    } else {
      WAIT_ASYNC(0);
    }
    __syncthreads();

    // ---- fragments + 8 WMMAs --------------------------------------------
    AB a[2];
#pragma unroll
    for (int t = 0; t < 2; ++t) {
      const _Float16* p = &sA[cur][(wm * 32 + t * 16 + l16) * 32 + hi * 8];
      a[t].h[0] = *(const v8h*)p;
      a[t].h[1] = *(const v8h*)(p + 16);
    }
    v16h bfr[4];
#pragma unroll
    for (int u = 0; u < 4; ++u)
      bfr[u] = *(const v16h*)(&sB[cur][(wn * 64 + u * 16 + l16) * 32 + hi * 16]), bfr[u].lo = *(const v8h*)(&sB[cur][(wn * 64 + u * 16 + l16) * 32 + hi * 8]), bfr[u].hi = *(const v8h*)(&sB[cur][(wn * 64 + u * 16 + l16) * 32 + hi * 8 + 16]);

#pragma unroll
    for (int t = 0; t < 2; ++t)
#pragma unroll
      for (int u = 0; u < 4; ++u) acc[t][u] = wmma_f16(a[t].v, bfr[u], acc[t][u]);

    __syncthreads();   // protect buffer reuse before next-stage writes
  }

  // ---- epilogue: bias + scatter into mode-specific layout ---------------
#pragma unroll
  for (int u = 0; u < 4; ++u) {
    const int n  = n0 + wn * 64 + u * 16 + l16;
    const float bv = bias[n];
    const int hh = n >> 6;       // head
    const int dd = n & 63;       // dim within head
#pragma unroll
    for (int t = 0; t < 2; ++t) {
#pragma unroll
      for (int i = 0; i < 8; ++i) {
        const int m = m0 + wm * 32 + t * 16 + i + hi * 8;
        const float val = acc[t][u][i] + bv;
        if (MODE == 3) {
          ((float*)Out)[(size_t)m * D_MODEL + n] = val;
        } else {
          const int b_ = m >> 13;
          const int r  = m & 8191;
          const int g  = r >> 8;
          const int c  = r & 255;
          size_t addr;
          if (MODE == 2)  // V transposed: [B,H,G,hd,C]
            addr = (((size_t)((b_ * NHEADS + hh) * GBLK + g) * HD + dd) * CHUNK + c);
          else            // Q / K: [B,H,G,C,hd]
            addr = (((size_t)((b_ * NHEADS + hh) * GBLK + g) * CHUNK + c) * HD + dd);
          ((_Float16*)Out)[addr] = (_Float16)val;
        }
      }
    }
  }
}

// ---------------------------------------------------------------------------
// Flash-style block attention. One workgroup = 8 waves = 128 query rows of
// one (b,h,g) chunk; grid = 2048 chunks * 2 halves. Each wave owns 16 rows.
// ---------------------------------------------------------------------------
__global__ __launch_bounds__(256) void attn_kernel(const _Float16* __restrict__ Q,
                                                   const _Float16* __restrict__ K,
                                                   const _Float16* __restrict__ Vt,
                                                   _Float16* __restrict__ ctx) {
  __shared__ _Float16 scratch[8 * 512];   // per-wave 16x32 f16 P tile

  const int tid  = threadIdx.x;
  const int lane = tid & 31;
  const int w    = tid >> 5;
  const int hi   = lane >> 4;
  const int l16  = lane & 15;

  const int idx     = blockIdx.x;
  const int halfIdx = idx & 1;
  const int bhg     = idx >> 1;
  const int g = bhg & 31;
  const int h = (bhg >> 5) & 15;
  const int b = bhg >> 9;

  const _Float16* Qb = Q  + (size_t)bhg * CHUNK * HD;
  const _Float16* Kb = K  + (size_t)bhg * CHUNK * HD;
  const _Float16* Vb = Vt + (size_t)bhg * HD * CHUNK;

  const int rowbase = halfIdx * 128 + w * 16;

  // Q A-fragments (held for the whole K loop), loaded straight from global.
  AB aq[2];
#pragma unroll
  for (int dh = 0; dh < 2; ++dh) {
    const _Float16* p = Qb + (size_t)(rowbase + l16) * HD + dh * 32 + hi * 8;
    aq[dh].h[0] = *(const v8h*)p;
    aq[dh].h[1] = *(const v8h*)(p + 16);
  }

  v8f o[4];
#pragma unroll
  for (int u = 0; u < 4; ++u) o[u] = zero8();
  float mi[8], li[8];
#pragma unroll
  for (int i = 0; i < 8; ++i) { mi[i] = -1e30f; li[i] = 0.f; }

  _Float16* sc = scratch + w * 512;
  const float scale = 0.03125f;  // 1/sqrt(D_MODEL)

  for (int kb = 0; kb < 8; ++kb) {      // 8 blocks of 32 keys
    // ---- S = Q @ K^T for 16 rows x 32 keys ------------------------------
    v8f s[2] = {zero8(), zero8()};
#pragma unroll
    for (int t = 0; t < 2; ++t) {
#pragma unroll
      for (int dh = 0; dh < 2; ++dh) {
        AB bk_;
        bk_.h[0] = *(const v8h*)(Kb + (size_t)(kb * 32 + t * 16 + l16) * HD + dh * 32 + hi * 8);
        bk_.h[1] = *(const v8h*)(Kb + (size_t)(kb * 32 + t * 16 + l16) * HD + dh * 32 + hi * 8 + 16);
        v16h bk = bk_.v;
        s[t] = wmma_f16(aq[dh].v, bk, s[t]);
      }
    }
    // ---- online softmax: rows align with C-fragment M mapping -----------
#pragma unroll
    for (int i = 0; i < 8; ++i) {
      float v0 = s[0][i] * scale;
      float v1 = s[1][i] * scale;
      float vmax = fmaxf(v0, v1);
#pragma unroll
      for (int off = 8; off >= 1; off >>= 1)
        vmax = fmaxf(vmax, __shfl_xor(vmax, off, 16));
      const float newm  = fmaxf(mi[i], vmax);
      const float alpha = __expf(mi[i] - newm);
      mi[i] = newm;
      const float p0 = __expf(v0 - newm);
      const float p1 = __expf(v1 - newm);
      float rs = p0 + p1;
#pragma unroll
      for (int off = 8; off >= 1; off >>= 1) rs += __shfl_xor(rs, off, 16);
      li[i] = li[i] * alpha + rs;
#pragma unroll
      for (int u = 0; u < 4; ++u) o[u][i] *= alpha;
      // stash P in row-major 16x32 scratch (C layout -> A layout transpose)
      const int m = i + hi * 8;
      sc[m * 32 + l16]      = (_Float16)p0;   // key tile 0
      sc[m * 32 + 16 + l16] = (_Float16)p1;   // key tile 1
    }
    __syncthreads();

    // ---- O += P @ V  (4 d-tiles of 16) ----------------------------------
    AB ap;
    {
      const _Float16* p = sc + l16 * 32 + hi * 8;
      ap.h[0] = *(const v8h*)p;
      ap.h[1] = *(const v8h*)(p + 16);
    }
#pragma unroll
    for (int u = 0; u < 4; ++u) {
      AB bv_;
      bv_.h[0] = *(const v8h*)(Vb + (size_t)(u * 16 + l16) * CHUNK + kb * 32 + hi * 8);
      bv_.h[1] = *(const v8h*)(Vb + (size_t)(u * 16 + l16) * CHUNK + kb * 32 + hi * 8 + 16);
      v16h bv = bv_.v;
      o[u] = wmma_f16(ap.v, bv, o[u]);
    }
    __syncthreads();
  }

  // ---- epilogue: O / l -> ctx f16 [B,N,D] --------------------------------
#pragma unroll
  for (int i = 0; i < 8; ++i) {
    const float inv  = 1.f / li[i];
    const int   mrow = b * SEQ + g * CHUNK + rowbase + i + hi * 8;
#pragma unroll
    for (int u = 0; u < 4; ++u) {
      const int col = h * HD + u * 16 + l16;
      ctx[(size_t)mrow * D_MODEL + col] = (_Float16)(o[u][i] * inv);
    }
  }
}

// ---------------------------------------------------------------------------
extern "C" void kernel_launch(void* const* d_in, const int* in_sizes, int n_in,
                              void* d_out, int out_size, void* d_ws, size_t ws_size,
                              hipStream_t stream) {
  const float* x  = (const float*)d_in[0];
  const float* Wq = (const float*)d_in[1];
  const float* bq = (const float*)d_in[2];
  const float* Wk = (const float*)d_in[3];
  const float* bk = (const float*)d_in[4];
  const float* Wv = (const float*)d_in[5];
  const float* bv = (const float*)d_in[6];
  const float* Wo = (const float*)d_in[7];
  const float* bo = (const float*)d_in[8];

  const size_t WELEM = (size_t)D_MODEL * D_MODEL;          // 1M halves
  const size_t BND   = (size_t)BATCH * SEQ * D_MODEL;      // 33.5M halves

  _Float16* WqT = (_Float16*)d_ws;
  _Float16* WkT = WqT + WELEM;
  _Float16* WvT = WkT + WELEM;
  _Float16* WoT = WvT + WELEM;
  _Float16* xh  = WoT + WELEM;
  _Float16* Qb  = xh + BND;
  _Float16* Kb  = Qb + BND;
  _Float16* Vtb = Kb + BND;
  _Float16* ctx = Vtb + BND;

  const dim3 tb(32, 8), tg(32, 32);
  transpose_f16<<<tg, tb, 0, stream>>>(Wq, WqT);
  transpose_f16<<<tg, tb, 0, stream>>>(Wk, WkT);
  transpose_f16<<<tg, tb, 0, stream>>>(Wv, WvT);
  transpose_f16<<<tg, tb, 0, stream>>>(Wo, WoT);

  convert_f16<<<(int)(BND / (8 * 256)), 256, 0, stream>>>(x, xh);  // 16384 blocks

  const dim3 gg(D_MODEL / 128, (BATCH * SEQ) / 128);       // (8, 256)
  wmma_gemm<0><<<gg, 256, 0, stream>>>(xh, WqT, bq, Qb);
  wmma_gemm<0><<<gg, 256, 0, stream>>>(xh, WkT, bk, Kb);
  wmma_gemm<2><<<gg, 256, 0, stream>>>(xh, WvT, bv, Vtb);

  attn_kernel<<<BATCH * NHEADS * GBLK * 2, 256, 0, stream>>>(Qb, Kb, Vtb, ctx);

  wmma_gemm<3><<<gg, 256, 0, stream>>>(ctx, WoT, bo, (float*)d_out);
}
